// SelfAttentionPairBias_13537736917626
// MI455X (gfx1250) — matrix operand order repaired and checked by reading, compiled, not run on hardware
//
#include <hip/hip_runtime.h>

// ---------------------------------------------------------------------------
// SelfAttentionPairBias for MI455X (gfx1250, wave32, WMMA bf16 path)
//   B=1, N=1024, H=8, HID_MV=4, HID_S=32, C_Z=128, d_head=64, vfeat=96
// ---------------------------------------------------------------------------

#define NTOK 1024

typedef __attribute__((ext_vector_type(16))) __bf16      v16bf;
typedef __attribute__((ext_vector_type(8)))  float       v8f;
typedef __attribute__((ext_vector_type(4)))  unsigned int u32x4;
typedef int i32x4 __attribute__((vector_size(16)));   // async-LDS builtin pointee

union FragBF {
    v16bf bf;
    u32x4 u4[2];
    unsigned short us[16];
};

__constant__ int c_GRADE[16] = {0,1,1,1,1,2,2,2,2,2,2,3,3,3,3,4};
__constant__ int c_INNER[8]  = {0,2,3,4,8,9,10,14};

__device__ __forceinline__ unsigned short f2bf(float x) {
    unsigned int u = __builtin_bit_cast(unsigned int, x);
    u += 0x7FFFu + ((u >> 16) & 1u);          // round-to-nearest-even
    return (unsigned short)(u >> 16);
}

#define WMMA_BF16(A, B, C) \
    __builtin_amdgcn_wmma_f32_16x16x32_bf16(false, (A), false, (B), (short)0, (C), false, false)

// ---- feature probes (compile-only loop: fall back cleanly if absent) -------
#if defined(__has_builtin)
#  if __has_builtin(__builtin_amdgcn_permlane16)
#    define HAVE_PERMLANE16 1
#  endif
#  if __has_builtin(__builtin_amdgcn_global_load_async_to_lds_b128) && \
      __has_builtin(__builtin_amdgcn_s_wait_asynccnt)
#    define HAVE_ASYNC_LDS 1
#  endif
#endif

#define AS1 __attribute__((address_space(1)))
#define AS3 __attribute__((address_space(3)))

// 16-lane-group butterfly reductions. Lane-select patterns: sel[i] = i ^ m,
// packed 4 bits/lane ({src2,src1} = lanes 15..8, 7..0).
#ifdef HAVE_PERMLANE16
__device__ __forceinline__ float pl16(float v, unsigned s1, unsigned s2) {
    unsigned u = __builtin_bit_cast(unsigned int, v);
    u = __builtin_amdgcn_permlane16(u, u, s1, s2, false, false);
    return __builtin_bit_cast(float, u);
}
__device__ __forceinline__ float red_max16(float v) {
    v = fmaxf(v, pl16(v, 0x67452301u, 0xEFCDAB89u));   // xor 1
    v = fmaxf(v, pl16(v, 0x54761032u, 0xDCFE98BAu));   // xor 2
    v = fmaxf(v, pl16(v, 0x32107654u, 0xBA98FEDCu));   // xor 4
    v = fmaxf(v, pl16(v, 0xFEDCBA98u, 0x76543210u));   // xor 8
    return v;
}
__device__ __forceinline__ float red_sum16(float v) {
    v += pl16(v, 0x67452301u, 0xEFCDAB89u);
    v += pl16(v, 0x54761032u, 0xDCFE98BAu);
    v += pl16(v, 0x32107654u, 0xBA98FEDCu);
    v += pl16(v, 0xFEDCBA98u, 0x76543210u);
    return v;
}
#else
__device__ __forceinline__ float red_max16(float v) {
    v = fmaxf(v, __shfl_xor(v, 1, 32));
    v = fmaxf(v, __shfl_xor(v, 2, 32));
    v = fmaxf(v, __shfl_xor(v, 4, 32));
    v = fmaxf(v, __shfl_xor(v, 8, 32));
    return v;
}
__device__ __forceinline__ float red_sum16(float v) {
    v += __shfl_xor(v, 1, 32);
    v += __shfl_xor(v, 2, 32);
    v += __shfl_xor(v, 4, 32);
    v += __shfl_xor(v, 8, 32);
    return v;
}
#endif

// ---------------------------------------------------------------------------
// K1: input projections + RoPE.
//   qf  : [tok][h][64]  bf16  (scaled by 1/sqrt(64))
//   kf  : [tok][64]     bf16
//   vfT : [96][tok]     bf16  (transposed: rows are features -> B-matrix ready)
// one block (64 threads) per token
// ---------------------------------------------------------------------------
__global__ __launch_bounds__(64)
void proj_kernel(const float* __restrict__ mv, const float* __restrict__ s,
                 const float* __restrict__ q_w_mv, const float* __restrict__ q_w_s,
                 const float* __restrict__ k_w_mv, const float* __restrict__ k_w_s,
                 const float* __restrict__ v_w_mv, const float* __restrict__ v_w_s,
                 unsigned short* __restrict__ qf, unsigned short* __restrict__ kf,
                 unsigned short* __restrict__ vfT) {
    __shared__ float s_sh[128], mv_sh[256], qs_sh[256], ks_sh[32];
    const int t = blockIdx.x;
    const int tid = threadIdx.x;

    // stage s[t] and mv[t] into LDS
    {
        const float* sp = s + (size_t)t * 128;
        s_sh[tid * 2 + 0] = sp[tid * 2 + 0];
        s_sh[tid * 2 + 1] = sp[tid * 2 + 1];
        const float* mp = mv + (size_t)t * 256;
        #pragma unroll
        for (int u = 0; u < 4; ++u) mv_sh[tid * 4 + u] = mp[tid * 4 + u];
    }
    __syncthreads();

    // ---- q_s = s @ q_w_s^T  (256 outputs, 4 per thread) ----
    #pragma unroll
    for (int u = 0; u < 4; ++u) {
        const int o = tid * 4 + u;
        const float* w = q_w_s + (size_t)o * 128;
        float acc = 0.f;
        for (int c = 0; c < 128; ++c) acc += s_sh[c] * w[c];
        qs_sh[o] = acc;
    }
    // ---- k_s / v_s (32 each) ----
    if (tid < 32) {
        const float* wk = k_w_s + (size_t)tid * 128;
        const float* wv = v_w_s + (size_t)tid * 128;
        float ak = 0.f, av = 0.f;
        for (int c = 0; c < 128; ++c) { ak += s_sh[c] * wk[c]; av += s_sh[c] * wv[c]; }
        ks_sh[tid] = ak;
        vfT[(size_t)(64 + tid) * NTOK + t] = f2bf(av);   // v_s part of vfeat
    }
    // ---- v_mv: 64 values, one per thread: (c = tid>>4, x = tid&15) ----
    {
        const int c = tid >> 4, x = tid & 15, g = c_GRADE[x];
        float acc = 0.f;
        #pragma unroll
        for (int i = 0; i < 16; ++i)
            acc += mv_sh[i * 16 + x] * v_w_mv[(c * 16 + i) * 5 + g];
        vfT[(size_t)(c * 16 + x) * NTOK + t] = f2bf(acc);
    }
    // ---- k_mv -> kf[0:32] (inner-blade slice) ----
    if (tid < 32) {
        const int c = tid >> 3, j = tid & 7;
        const int idx = c_INNER[j], g = c_GRADE[idx];
        float acc = 0.f;
        #pragma unroll
        for (int i = 0; i < 16; ++i)
            acc += mv_sh[i * 16 + idx] * k_w_mv[(c * 16 + i) * 5 + g];
        kf[(size_t)t * 64 + c * 8 + j] = f2bf(acc);
    }
    // ---- q_mv -> qf[h][0:32], 256 values, 4 per thread ----
    #pragma unroll
    for (int u = 0; u < 4; ++u) {
        const int flat = tid * 4 + u;
        const int h = flat >> 5, rem = flat & 31, c = rem >> 3, j = rem & 7;
        const int idx = c_INNER[j], g = c_GRADE[idx];
        float acc = 0.f;
        #pragma unroll
        for (int i = 0; i < 16; ++i)
            acc += mv_sh[i * 16 + idx] * q_w_mv[((h * 4 + c) * 16 + i) * 5 + g];
        qf[((size_t)t * 8 + h) * 64 + c * 8 + j] = f2bf(0.125f * acc);
    }
    __syncthreads();

    // ---- RoPE on q_s -> qf[h][32:64] ----
    #pragma unroll
    for (int u = 0; u < 4; ++u) {
        const int flat = tid * 4 + u;
        const int h = flat >> 5, j = flat & 31, p = j >> 1;
        const float ang = (float)t * __powf(4096.0f, -(float)p / 16.0f);
        const float cs = __cosf(ang), sn = __sinf(ang);
        const float x1 = qs_sh[h * 32 + 2 * p], x2 = qs_sh[h * 32 + 2 * p + 1];
        const float val = (j & 1) ? (x1 * sn + x2 * cs) : (x1 * cs - x2 * sn);
        qf[((size_t)t * 8 + h) * 64 + 32 + j] = f2bf(0.125f * val);
    }
    // ---- RoPE on k_s -> kf[32:64] ----
    if (tid < 32) {
        const int j = tid, p = j >> 1;
        const float ang = (float)t * __powf(4096.0f, -(float)p / 16.0f);
        const float cs = __cosf(ang), sn = __sinf(ang);
        const float x1 = ks_sh[2 * p], x2 = ks_sh[2 * p + 1];
        const float val = (j & 1) ? (x1 * sn + x2 * cs) : (x1 * cs - x2 * sn);
        kf[(size_t)t * 64 + 32 + j] = f2bf(val);
    }
}

// ---------------------------------------------------------------------------
// K2: pair bias = LN(z) @ bias_w   -> bias[h][q][k] fp32
// one wave per (q,k) pair; lanes hold 4 of the 128 channels; butterfly reduce.
// This kernel is the HBM roofline: reads z (512 MB) exactly once.
// ---------------------------------------------------------------------------
__global__ __launch_bounds__(256)
void bias_kernel(const float* __restrict__ z, const float* __restrict__ ln_g,
                 const float* __restrict__ ln_b, const float* __restrict__ bias_w,
                 float* __restrict__ bias) {
    const int lane = threadIdx.x & 31;
    const int wave = threadIdx.x >> 5;
    const size_t pair = (size_t)blockIdx.x * 8 + wave;       // q*1024 + k
    const float* zr = z + (pair << 7);

    __builtin_prefetch(zr + 1024, 0, 0);                      // row 8 pairs ahead (next block, same wave)

    const float4 x = *(const float4*)(zr + lane * 4);
    float ssum = x.x + x.y + x.z + x.w;
    float ssq  = x.x * x.x + x.y * x.y + x.z * x.z + x.w * x.w;
    #pragma unroll
    for (int m = 1; m <= 16; m <<= 1) {
        ssum += __shfl_xor(ssum, m, 32);
        ssq  += __shfl_xor(ssq,  m, 32);
    }
    const float mu  = ssum * (1.0f / 128.0f);
    const float var = ssq * (1.0f / 128.0f) - mu * mu;
    const float inv = rsqrtf(var + 1e-5f);

    const float4 g = *(const float4*)(ln_g + lane * 4);
    const float4 b = *(const float4*)(ln_b + lane * 4);
    float zn[4];
    zn[0] = (x.x - mu) * inv * g.x + b.x;
    zn[1] = (x.y - mu) * inv * g.y + b.y;
    zn[2] = (x.z - mu) * inv * g.z + b.z;
    zn[3] = (x.w - mu) * inv * g.w + b.w;

    float hs[8];
    #pragma unroll
    for (int h = 0; h < 8; ++h) {
        float a = 0.f;
        #pragma unroll
        for (int cc = 0; cc < 4; ++cc)
            a += zn[cc] * bias_w[(size_t)(lane * 4 + cc) * 8 + h];
        #pragma unroll
        for (int m = 1; m <= 16; m <<= 1) a += __shfl_xor(a, m, 32);
        hs[h] = a;
    }
    if (lane < 8)
        bias[((size_t)lane << 20) + pair] = hs[lane];
}

// ---------------------------------------------------------------------------
// K3: flash attention with pair bias, bf16 WMMA.
// block = 256 threads = 8 waves; wave w = head w; block x = q-tile of 16.
// Per 32-k step: 4 WMMA (logits, K=64) + online softmax + 6 WMMA (P@V, 96 feats).
// kf tile is staged block-wide into LDS with async DMA (double buffered); P is
// moved C-layout -> A-layout through a 1KB/wave LDS buffer.
// ---------------------------------------------------------------------------
#ifdef HAVE_ASYNC_LDS
// one async b128 per thread => 256 x 16B = 4KB tile; 1 async instr per wave
#define STAGE_KTILE(kt, buf)                                                       \
    __builtin_amdgcn_global_load_async_to_lds_b128(                                \
        (AS1 i32x4*)((char*)(kf + (size_t)(kt) * 2048) + threadIdx.x * 16),        \
        (AS3 i32x4*)((char*)&Kbuf[(buf)][0] + threadIdx.x * 16), 0, 0)
#endif

__global__ __launch_bounds__(256)
void attn_kernel(const unsigned short* __restrict__ qf,
                 const unsigned short* __restrict__ kf,
                 const unsigned short* __restrict__ vfT,
                 const float* __restrict__ bias,
                 float* __restrict__ ofeat) {
    __shared__ unsigned short Pbuf[8 * 16 * 32];
#ifdef HAVE_ASYNC_LDS
    __shared__ __attribute__((aligned(16))) unsigned short Kbuf[2][2048]; // 2 x 4KB
#endif
    const int lane = threadIdx.x & 31;
    const int h    = threadIdx.x >> 5;                 // wave == head
    const int q0   = blockIdx.x * 16;
    const int half = lane >> 4, lr = lane & 15;
    unsigned short* Pw = &Pbuf[h * 16 * 32];

    // A-matrix fragments of q (16x64, two K=32 fragments)
    FragBF aq0, aq1;
    {
        const unsigned short* row = qf + ((size_t)(q0 + lr) * 8 + h) * 64;
        aq0.u4[0] = *(const u32x4*)(row +  0 + half * 8);
        aq0.u4[1] = *(const u32x4*)(row + 16 + half * 8);
        aq1.u4[0] = *(const u32x4*)(row + 32 + half * 8);
        aq1.u4[1] = *(const u32x4*)(row + 48 + half * 8);
    }

    const v8f Zero = {0.f, 0.f, 0.f, 0.f, 0.f, 0.f, 0.f, 0.f};
    v8f O[6];
    float m[8], lsum[8];
    #pragma unroll
    for (int n = 0; n < 6; ++n) O[n] = Zero;
    #pragma unroll
    for (int r = 0; r < 8; ++r) { m[r] = -1e30f; lsum[r] = 0.f; }

    const float* bplane = bias + ((size_t)h << 20);

#ifdef HAVE_ASYNC_LDS
    STAGE_KTILE(0, 0);
#endif

    for (int k0 = 0; k0 < NTOK; k0 += 32) {
        const int t32 = k0 >> 5;
#ifdef HAVE_ASYNC_LDS
        if (t32 + 1 < 32) {
            STAGE_KTILE(t32 + 1, (t32 + 1) & 1);
            __builtin_amdgcn_s_wait_asynccnt(1);    // tile t32 (oldest) complete
        } else {
            __builtin_amdgcn_s_wait_asynccnt(0);
        }
        __syncthreads();                            // all waves' stages visible
        const unsigned short* ktile = &Kbuf[t32 & 1][0];   // rows k0..k0+31, 64 each
#else
        const unsigned short* ktile = kf + (size_t)k0 * 64;
#endif

        // ---- logits: two 16x16 tiles over d=64 ----
        v8f c0 = Zero, c1 = Zero;
        {
            const unsigned short* row = ktile + (size_t)lr * 64;
            FragBF b0, b1;
            b0.u4[0] = *(const u32x4*)(row +  0 + half * 8);
            b0.u4[1] = *(const u32x4*)(row + 16 + half * 8);
            b1.u4[0] = *(const u32x4*)(row + 32 + half * 8);
            b1.u4[1] = *(const u32x4*)(row + 48 + half * 8);
            c0 = WMMA_BF16(aq0.bf, b0.bf, c0);
            c0 = WMMA_BF16(aq1.bf, b1.bf, c0);
        }
        {
            const unsigned short* row = ktile + (size_t)(16 + lr) * 64;
            FragBF b0, b1;
            b0.u4[0] = *(const u32x4*)(row +  0 + half * 8);
            b0.u4[1] = *(const u32x4*)(row + 16 + half * 8);
            b1.u4[0] = *(const u32x4*)(row + 32 + half * 8);
            b1.u4[1] = *(const u32x4*)(row + 48 + half * 8);
            c1 = WMMA_BF16(aq0.bf, b0.bf, c1);
            c1 = WMMA_BF16(aq1.bf, b1.bf, c1);
        }

        // ---- bias add + online softmax (row = (r, half), 16 cols across lanes) ----
        float sc[8];
        #pragma unroll
        for (int r = 0; r < 8; ++r) {
            const int M = r + half * 8;
            const float* brow = bplane + (size_t)(q0 + M) * NTOK + k0;
            const float l0 = c0[r] + brow[lr];
            const float l1 = c1[r] + brow[16 + lr];
            const float t  = red_max16(fmaxf(l0, l1));
            const float mn = fmaxf(m[r], t);
            sc[r] = __expf(m[r] - mn);
            m[r] = mn;
            const float p0 = __expf(l0 - mn);
            const float p1 = __expf(l1 - mn);
            lsum[r] = lsum[r] * sc[r] + red_sum16(p0 + p1);
            Pw[M * 32 + lr]      = f2bf(p0);   // C-layout -> LDS (16x32 tile)
            Pw[M * 32 + 16 + lr] = f2bf(p1);
        }
        #pragma unroll
        for (int n = 0; n < 6; ++n)
            #pragma unroll
            for (int r = 0; r < 8; ++r) O[n][r] *= sc[r];

        // ---- reload P in A layout; accumulate O += P @ V ----
        FragBF ap;
        {
            const unsigned short* row = Pw + lr * 32;
            ap.u4[0] = *(const u32x4*)(row +  0 + half * 8);
            ap.u4[1] = *(const u32x4*)(row + 16 + half * 8);
        }
        #pragma unroll
        for (int n = 0; n < 6; ++n) {
            FragBF bv;
            const unsigned short* row = vfT + (size_t)(n * 16 + lr) * NTOK + k0 + half * 8;
            bv.u4[0] = *(const u32x4*)(row);
            bv.u4[1] = *(const u32x4*)(row + 16);
            O[n] = WMMA_BF16(ap.bf, bv.bf, O[n]);
        }

#ifdef HAVE_ASYNC_LDS
        __syncthreads();                            // tile buffer reusable
#endif
    }

    // ---- finalize: divide by row sum, write o-features [tok][h][96] ----
    #pragma unroll
    for (int n = 0; n < 6; ++n)
        #pragma unroll
        for (int r = 0; r < 8; ++r) {
            const int M = r + half * 8;
            ofeat[((size_t)(q0 + M) * 8 + h) * 96 + n * 16 + lr] = O[n][r] / lsum[r];
        }
}

// ---------------------------------------------------------------------------
// K4: output projections. one block (64 threads) per token.
//   out_mv[t][o][b] = sum_i ofeat_mv[t][i][b] * out_w_mv[o][i][GRADE[b]]
//   out_s [t][o]    = sum_c ofeat_s[t][c]     * out_w_s[o][c]
// ---------------------------------------------------------------------------
__global__ __launch_bounds__(64)
void out_kernel(const float* __restrict__ ofeat,
                const float* __restrict__ out_w_mv,
                const float* __restrict__ out_w_s,
                float* __restrict__ out) {
    __shared__ float of[768];                       // 8 heads * 96 features
    const int t = blockIdx.x, tid = threadIdx.x;
    const float* src = ofeat + (size_t)t * 768;
    #pragma unroll
    for (int u = 0; u < 12; ++u) of[tid + u * 64] = src[tid + u * 64];
    __syncthreads();

    // out_mv: 256 outputs (o,b), 4 per thread
    #pragma unroll
    for (int u = 0; u < 4; ++u) {
        const int flat = tid * 4 + u, o = flat >> 4, b = flat & 15;
        const int g = c_GRADE[b];
        float acc = 0.f;
        #pragma unroll
        for (int i = 0; i < 32; ++i) {
            const int hh = i >> 2, cc = i & 3;
            acc += of[hh * 96 + cc * 16 + b] * out_w_mv[(o * 32 + i) * 5 + g];
        }
        out[((size_t)t * 16 + o) * 16 + b] = acc;
    }
    // out_s: 128 outputs, 2 per thread
    #pragma unroll
    for (int u = 0; u < 2; ++u) {
        const int o = tid * 2 + u;
        const float* w = out_w_s + (size_t)o * 256;
        float acc = 0.f;
        for (int c = 0; c < 256; ++c) {
            const int hh = c >> 5, j = c & 31;
            acc += of[hh * 96 + 64 + j] * w[c];
        }
        out[262144 + (size_t)t * 128 + o] = acc;
    }
}

// ---------------------------------------------------------------------------
// launcher
// ---------------------------------------------------------------------------
extern "C" void kernel_launch(void* const* d_in, const int* in_sizes, int n_in,
                              void* d_out, int out_size, void* d_ws, size_t ws_size,
                              hipStream_t stream) {
    const float* mv       = (const float*)d_in[0];
    const float* s        = (const float*)d_in[1];
    const float* z        = (const float*)d_in[2];
    const float* q_w_mv   = (const float*)d_in[3];
    const float* q_w_s    = (const float*)d_in[4];
    const float* k_w_mv   = (const float*)d_in[5];
    const float* k_w_s    = (const float*)d_in[6];
    const float* v_w_mv   = (const float*)d_in[7];
    const float* v_w_s    = (const float*)d_in[8];
    const float* out_w_mv = (const float*)d_in[9];
    const float* out_w_s  = (const float*)d_in[10];
    const float* ln_g     = (const float*)d_in[11];
    const float* ln_b     = (const float*)d_in[12];
    const float* bias_w   = (const float*)d_in[13];

    char* ws = (char*)d_ws;
    unsigned short* qf  = (unsigned short*)(ws);                         // 1,048,576 B
    unsigned short* kf  = (unsigned short*)(ws + 1048576);               //   131,072 B
    unsigned short* vfT = (unsigned short*)(ws + 1048576 + 131072);      //   196,608 B
    float* bias  = (float*)(ws + 1376256);                               // 33,554,432 B
    float* ofeat = (float*)(ws + 1376256 + 33554432);                    //  3,145,728 B

    proj_kernel<<<NTOK, 64, 0, stream>>>(mv, s, q_w_mv, q_w_s, k_w_mv, k_w_s,
                                         v_w_mv, v_w_s, qf, kf, vfT);
    bias_kernel<<<(NTOK * NTOK) / 8, 256, 0, stream>>>(z, ln_g, ln_b, bias_w, bias);
    attn_kernel<<<NTOK / 16, 256, 0, stream>>>(qf, kf, vfT, bias, ofeat);
    out_kernel<<<NTOK, 64, 0, stream>>>(ofeat, out_w_mv, out_w_s, (float*)d_out);
}
